// TransitionGNN_18141941858916
// MI455X (gfx1250) — matrix operand order repaired and checked by reading, not compiled, not run on hardware
//
#include <hip/hip_runtime.h>
#include <hip/hip_bf16.h>

// ---------------------------------------------------------------------------
// TransitionGNN on MI455X (gfx1250, wave32, WMMA)
//
// Factorization: concat(sender,receiver) @ W1^T == sender@W1a^T + receiver@W1b^T
//   Stage 1 (cold): P[node] = [states@W1a^T + b1 | states@W1b^T], stored as
//                   bf16 PRE-SWIZZLED into WMMA-A element order (4 MB, L2-resident)
//   Stage 2 (hot):  per edge (r->c): A-operand = pk_max(pk_add(Pa,Pb), 0)
//                   (packed bf16 add + relu, zero conversions), then
//                   out = sigmoid(H @ W2^T + b2) via v_wmma_f32_16x16x32_bf16.
// FLOPs: 38 GF naive -> ~4.8 GF; compute-bound on bf16 WMMA, traffic L2-bound.
// All f32->bf16 packing uses hardware v_cvt_pk_bf16_f32 (1 VALU op / 2 elems).
// ---------------------------------------------------------------------------

typedef __attribute__((ext_vector_type(16))) __bf16          v16bf;
typedef __attribute__((ext_vector_type(8)))  float           v8f;
typedef __attribute__((ext_vector_type(8)))  unsigned int    v8u;

static constexpr int BATCH = 64, NAG = 64;
static constexpr int D = 128, TWO_D = 256, HID = 256, OUT = 32;
static constexpr int NODES = BATCH * NAG;        // 4096
static constexpr int PCOLS = 512;                // [Pa(256) | Pb(256)] packed bf16
static constexpr size_t P_BYTES = (size_t)NODES * PCOLS * sizeof(unsigned short); // 4 MB

// hardware packed f32x2 -> bf16x2 (RNE), one VALU instruction
__device__ __forceinline__ unsigned int cvt_pk_bf16(float x, float y) {
    unsigned int r;
    asm("v_cvt_pk_bf16_f32 %0, %1, %2" : "=v"(r) : "v"(x), "v"(y));
    return r;
}

// WMMA-A packed position for feature f (within a 256-feature section):
//   lane-group hi, element i <-> f = (f>>5)*32 + (i<8?i:i+8) + 8*hi
__device__ __forceinline__ int apack(int f) {
    const int r = f & 31;
    return (f >> 5) * 32 + ((r >> 3) & 1) * 16 + (r & 7) + ((r >> 4) << 3);
}

// ---------------------------------------------------------------------------
// Stage 1: P = states @ [W1a;W1b]^T (M=4096, N=512, K=128), b1 folded into Pa,
// result stored as bf16 in WMMA-A swizzled order. One 16x16 tile per wave.
// ---------------------------------------------------------------------------
__global__ void __launch_bounds__(256)
proj_kernel(const float* __restrict__ states, const float* __restrict__ W1,
            const float* __restrict__ b1, unsigned short* __restrict__ Ppk)
{
    const int wave  = threadIdx.x >> 5;
    const int lane  = threadIdx.x & 31;
    const int tile  = blockIdx.x * 8 + wave;     // 8192 tiles = 256 M x 32 N
    const int mtile = tile >> 5;
    const int ntile = tile & 31;
    const int hi    = lane >> 4;
    const int lm    = lane & 15;

    const int m_a = mtile * 16 + lm;             // node row for A
    const int f_n = ntile * 16 + lm;             // output feature col (B/C lane)

    const int    w1row = (f_n < HID) ? f_n : (f_n - HID);
    const int    koff  = (f_n < HID) ? 0 : D;    // Pa: W1[:,:128]  Pb: W1[:,128:]
    const float* wp    = W1 + (size_t)w1row * TWO_D + koff;
    const float* ap    = states + (size_t)m_a * D;

    v8f acc = {};
    #pragma unroll
    for (int kb = 0; kb < D; kb += 32) {
        v8u au, bu;
        #pragma unroll
        for (int v = 0; v < 8; ++v) {
            const int ka = kb + 2 * v + ((v >= 4) ? 8 : 0) + 8 * hi;
            const float2 a2 = *(const float2*)(ap + ka);
            au[v] = cvt_pk_bf16(a2.x, a2.y);
            const int kw = kb + 2 * v + ((v >= 4) ? 8 : 0) + 8 * hi;
            const float2 w2 = *(const float2*)(wp + kw);
            bu[v] = cvt_pk_bf16(w2.x, w2.y);
        }
        const v16bf a = __builtin_bit_cast(v16bf, au);
        const v16bf b = __builtin_bit_cast(v16bf, bu);
        acc = __builtin_amdgcn_wmma_f32_16x16x32_bf16(
                  false, a, false, b, (short)0, acc, false, false);
    }

    const float bias = (f_n < HID) ? b1[f_n] : 0.0f;
    const int   isB  = (f_n >= HID);
    const int   p    = apack(isB ? (f_n - HID) : f_n) + (isB ? 256 : 0);
    #pragma unroll
    for (int v = 0; v < 8; ++v) {
        const int   m   = mtile * 16 + v + 8 * hi;   // C-layout row
        const float val = acc[v] + bias;
        Ppk[(size_t)m * PCOLS + p] = (unsigned short)cvt_pk_bf16(val, val);
    }
}

// ---------------------------------------------------------------------------
// Pre-swizzle W2 (32x256 f32) into WMMA-B bf16 layout: one aligned 32B load
// per operand in the hot loop. Index: ((kstep*2+ntile)*32 + lane)*16 bf16.
// ---------------------------------------------------------------------------
__global__ void __launch_bounds__(512)
packW2_kernel(const float* __restrict__ W2, unsigned int* __restrict__ Bp32)
{
    const int t     = threadIdx.x;       // 512 = 8 ksteps * 2 ntiles * 32 lanes
    const int kstep = t >> 6;
    const int ntile = (t >> 5) & 1;
    const int lane  = t & 31;
    const int hi = lane >> 4, lm = lane & 15;
    const int o  = ntile * 16 + lm;      // W2 row = B lane col
    const float* wp = W2 + (size_t)o * HID;
    unsigned int* dst = Bp32 + (size_t)t * 8;
    const int kb = kstep * 32;
    #pragma unroll
    for (int v = 0; v < 8; ++v) {
        const int k = kb + 2 * v + ((v >= 4) ? 8 : 0) + 8 * hi;
        const float2 w2 = *(const float2*)(wp + k);
        dst[v] = cvt_pk_bf16(w2.x, w2.y);
    }
}

// ---------------------------------------------------------------------------
// Stage 2 (hot): one workgroup (4 waves) per (batch, sender). Wave owns a
// 16-receiver M-tile. A-operand built with packed bf16 add+relu; 2 WMMAs per
// K-step (OUT=32 as two 16-col N-tiles). Epilogue: sigmoid + diagonal skip.
// ---------------------------------------------------------------------------
__global__ void __launch_bounds__(128)
edge_kernel(const unsigned short* __restrict__ Ppk,
            const unsigned short* __restrict__ Bp,
            const float* __restrict__ b2, float* __restrict__ out)
{
    const int wg    = blockIdx.x;        // 4096 = 64 batches * 64 senders
    const int b     = wg >> 6;
    const int r     = wg & 63;
    const int mtile = threadIdx.x >> 5;  // receiver tile 0..3
    const int lane  = threadIdx.x & 31;
    const int hi = lane >> 4, lm = lane & 15;

    const unsigned short* pa = Ppk + (size_t)(b * NAG + r) * PCOLS;         // Pa+b1
    const int             ca = mtile * 16 + lm;                             // A-row receiver
    const unsigned short* pb = Ppk + (size_t)(b * NAG + ca) * PCOLS + 256;  // Pb

    unsigned int zero = 0u;              // packed bf16 {+0,+0}
    v8f c0 = {}, c1 = {};
    #pragma unroll
    for (int ks = 0; ks < 8; ++ks) {
        const int off = ks * 32 + hi * 16;                 // swizzled, 32B step
        const v8u paw = *(const v8u*)(pa + off);           // sender (broadcast)
        const v8u pbw = *(const v8u*)(pb + off);           // receiver row
        v8u aw;
        #pragma unroll
        for (int j = 0; j < 8; ++j) {
            unsigned int s, h;
            // h = relu(Pa + Pb) on 2 bf16 lanes at once (VOP3P, CDNA5)
            asm("v_pk_add_bf16 %0, %1, %2"     : "=v"(s) : "v"(paw[j]), "v"(pbw[j]));
            asm("v_pk_max_num_bf16 %0, %1, %2" : "=v"(h) : "v"(s), "v"(zero));
            aw[j] = h;
        }
        const v16bf a   = __builtin_bit_cast(v16bf, aw);
        const v16bf b0  = __builtin_bit_cast(v16bf,
                              *(const v8u*)(Bp + (size_t)((ks * 2 + 0) * 32 + lane) * 16));
        const v16bf b1v = __builtin_bit_cast(v16bf,
                              *(const v8u*)(Bp + (size_t)((ks * 2 + 1) * 32 + lane) * 16));
        c0 = __builtin_amdgcn_wmma_f32_16x16x32_bf16(false, a, false, b0,  (short)0, c0, false, false);
        c1 = __builtin_amdgcn_wmma_f32_16x16x32_bf16(false, a, false, b1v, (short)0, c1, false, false);
    }

    const float bias0 = b2[lm];
    const float bias1 = b2[16 + lm];
    const size_t ebase = (size_t)(b * NAG + r) * 63;       // edges with sender r
    #pragma unroll
    for (int v = 0; v < 8; ++v) {
        const int crecv = mtile * 16 + v + 8 * hi;         // C-layout M row
        if (crecv == r) continue;                          // no self loops
        const int cadj = crecv - (crecv > r ? 1 : 0);
        float* op = out + (ebase + cadj) * OUT;
        op[lm]      = 1.f / (1.f + __expf(-(c0[v] + bias0)));
        op[16 + lm] = 1.f / (1.f + __expf(-(c1[v] + bias1)));
    }
}

// ---------------------------------------------------------------------------
extern "C" void kernel_launch(void* const* d_in, const int* in_sizes, int n_in,
                              void* d_out, int out_size, void* d_ws, size_t ws_size,
                              hipStream_t stream)
{
    const float* states = (const float*)d_in[0];   // [64,64,128]
    const float* W1     = (const float*)d_in[1];   // [256,256]
    const float* b1     = (const float*)d_in[2];   // [256]
    const float* W2     = (const float*)d_in[3];   // [32,256]
    const float* b2     = (const float*)d_in[4];   // [32]

    unsigned short* Ppk  = (unsigned short*)d_ws;                      // 4 MB
    void*           BpWs = (void*)((char*)d_ws + P_BYTES);             // 16 KB
    unsigned int*   Bp32 = (unsigned int*)BpWs;
    const unsigned short* Bp16 = (const unsigned short*)BpWs;

    packW2_kernel<<<1, 512, 0, stream>>>(W2, Bp32);
    proj_kernel<<<1024, 256, 0, stream>>>(states, W1, b1, Ppk);
    edge_kernel<<<4096, 128, 0, stream>>>(Ppk, Bp16, b2, (float*)d_out);
}
